// SemGCN_25314537243184
// MI455X (gfx1250) — matrix operand order repaired and checked by reading, not compiled, not run on hardware
//
#include <hip/hip_runtime.h>

typedef __attribute__((ext_vector_type(16))) _Float16 v16h;
typedef __attribute__((ext_vector_type(2)))  _Float16 v2h;
typedef __attribute__((ext_vector_type(8)))  float    v8f;

#define BB     4096
#define JJ     16
#define HIDD   128
#define INTERD 64
#define NLAY   4
#define NEGV   (-9.0e15f)
#define NROWS  (BB*JJ)          // 65536 rows for all BN stats

// ---------------- workspace layout (in floats) ----------------
constexpr size_t OFF_STATS = 0;                        // 14*256
constexpr size_t OFF_DIAG  = 4096;                     // 10*16
constexpr size_t OFF_OFFM  = 4352;                     // 10*256
constexpr size_t SZ_ACT    = (size_t)BB * JJ * HIDD;   // 8388608
constexpr size_t OFF_H     = 8192;
constexpr size_t OFF_T     = OFF_H  + SZ_ACT;
constexpr size_t OFF_H2    = OFF_T  + SZ_ACT;
constexpr size_t OFF_C     = OFF_H2 + SZ_ACT;          // B*16*192
constexpr size_t OFF_Y     = OFF_C  + (size_t)BB*JJ*192;
constexpr size_t OFF_WY    = OFF_Y  + (size_t)BB*JJ*INTERD;
constexpr size_t OFF_F16   = OFF_WY + SZ_ACT;          // f16 weight blobs start here
// f16 blob offsets (in halfs, relative to OFF_F16 base):
constexpr size_t WG_SZ = 8 * 512;     // per gconv k-step: 8 nf * 512
constexpr size_t WBLOB_GCONV  = 0;                         // 8 layers * 32768
constexpr size_t WBLOB_NLCONV = 8 * 32768;                 // 5 * 24576
constexpr size_t WBLOB_NLWY   = WBLOB_NLCONV + 5 * 24576;  // 5 * 8192

// ---- A-fragment swizzle mapping (16-bit A, 16x32 per k-step) ----
// (m, kk) -> lane = m + 16*g, element e:
//   g=kk>>4, e=kk&15
__device__ __forceinline__ void amap(int kk, int& lt, int& e) {
  lt = (kk >> 4) << 4; e = kk & 15;
}
// ---- B-fragment swizzle mapping (32x16 per k-step) ----
// (kk, nn) -> lane = (kk>>4)*16 + nn, e = kk & 15  (inverse used by swz kernels)

#define WMMA_F16(a, b, c) \
  __builtin_amdgcn_wmma_f32_16x16x32_f16(false, (a), false, (b), (short)0, (c), false, false)

// ---------------- weight pre-swizzle kernels (run once per launch) ----------------
__global__ void swz_gconv_w_kernel(const float* __restrict__ Wres, _Float16* __restrict__ dst) {
  int l = blockIdx.y;
  int idx = blockIdx.x * 256 + threadIdx.x;            // 0..32767
  int e = idx & 15, lane = (idx >> 4) & 31, n = (idx >> 9) & 7, ks = idx >> 12;
  int kk = ((lane >> 4) << 4) | e, nn = lane & 15;
  int K = ks * 32 + kk, o = n * 16 + nn;
  const float* Wl = Wres + (size_t)l * 2 * HIDD * HIDD;
  float v = (K < 128) ? Wl[(size_t)K * HIDD + o]
                      : Wl[(size_t)HIDD * HIDD + (size_t)(K - 128) * HIDD + o];
  dst[(size_t)l * 32768 + idx] = (_Float16)v;
}
__global__ void swz_nlconv_w_kernel(const float* __restrict__ gw, const float* __restrict__ tw,
                                    const float* __restrict__ pw, _Float16* __restrict__ dst) {
  int i = blockIdx.y;
  int idx = blockIdx.x * 256 + threadIdx.x;            // 0..24575
  int e = idx & 15, lane = (idx >> 4) & 31;
  int n = (idx >> 9) % 12, ks = idx / (12 * 512);
  int kk = ((lane >> 4) << 4) | e, nn = lane & 15;
  int K = ks * 32 + kk, oc = n * 16 + nn;
  size_t lw = (size_t)i * INTERD * HIDD;
  float v = (oc < 64)  ? gw[lw + (size_t)oc * HIDD + K]
          : (oc < 128) ? tw[lw + (size_t)(oc - 64) * HIDD + K]
                       : pw[lw + (size_t)(oc - 128) * HIDD + K];
  dst[(size_t)i * 24576 + idx] = (_Float16)v;
}
__global__ void swz_nlwy_w_kernel(const float* __restrict__ Wm, _Float16* __restrict__ dst) {
  int i = blockIdx.y;
  int idx = blockIdx.x * 256 + threadIdx.x;            // 0..8191
  int e = idx & 15, lane = (idx >> 4) & 31, n = (idx >> 9) & 7, ks = idx >> 12;
  int kk = ((lane >> 4) << 4) | e, nn = lane & 15;
  int K = ks * 32 + kk, o = n * 16 + nn;
  float v = Wm[(size_t)i * HIDD * INTERD + (size_t)o * INTERD + K];
  dst[(size_t)i * 8192 + idx] = (_Float16)v;
}

// ---------------- small utility kernels ----------------
__global__ void zero_kernel(float* p, int n) {
  int i = blockIdx.x * blockDim.x + threadIdx.x;
  if (i < n) p[i] = 0.f;
}

__global__ void prep_adj_kernel(const float* __restrict__ gc_in_e,
                                const float* __restrict__ e_res,
                                const float* __restrict__ gc_out_e,
                                const int* __restrict__ rows, const int* __restrict__ cols,
                                int nnz, float* __restrict__ diag_out,
                                float* __restrict__ off_out) {
  int tid = blockIdx.x * blockDim.x + threadIdx.x;
  if (tid >= 160) return;
  int l = tid / 16, r = tid % 16;
  const float* e = (l == 0) ? gc_in_e : (l <= 8 ? e_res + (size_t)(l - 1) * nnz : gc_out_e);
  float logit[16];
#pragma unroll
  for (int c = 0; c < 16; ++c) logit[c] = NEGV;
  for (int n = 0; n < nnz; ++n)
    if (rows[n] == r) logit[cols[n]] = e[n];
  float mx = logit[0];
#pragma unroll
  for (int c = 1; c < 16; ++c) mx = fmaxf(mx, logit[c]);
  float ex[16], s = 0.f;
#pragma unroll
  for (int c = 0; c < 16; ++c) { ex[c] = __expf(logit[c] - mx); s += ex[c]; }
  float inv = 1.f / s;
#pragma unroll
  for (int c = 0; c < 16; ++c) {
    float a = ex[c] * inv;
    off_out[l * 256 + r * 16 + c] = (c == r) ? 0.f : a;
    if (c == r) diag_out[l * 16 + r] = a;
  }
}

// Input gconv: K=2, pure VALU.
__global__ __launch_bounds__(128) void gc_in_kernel(
    const float* __restrict__ x, const float* __restrict__ W /*(2,2,128)*/,
    const float* __restrict__ bias, const float* __restrict__ diag,
    const float* __restrict__ offm, float* __restrict__ t, float* __restrict__ stats) {
  __shared__ float xs[16][2];
  __shared__ float soff[16][16];
  __shared__ float sdiag[16];
  int b = blockIdx.x, tid = threadIdx.x;
  if (tid < 32) xs[tid >> 1][tid & 1] = x[(size_t)b * 32 + tid];
  if (tid < 16) sdiag[tid] = diag[tid];
  for (int i = tid; i < 256; i += 128) soff[i >> 4][i & 15] = offm[i];
  __syncthreads();
  int o = tid;
  float w00 = W[o], w01 = W[128 + o], w10 = W[256 + o], w11 = W[384 + o];
  float h0[16], h1[16];
#pragma unroll
  for (int j = 0; j < 16; ++j) {
    h0[j] = xs[j][0] * w00 + xs[j][1] * w01;
    h1[j] = xs[j][0] * w10 + xs[j][1] * w11;
  }
  float bo = bias[o], s = 0.f, s2 = 0.f;
#pragma unroll
  for (int j = 0; j < 16; ++j) {
    float v = sdiag[j] * h0[j] + bo;
#pragma unroll
    for (int j2 = 0; j2 < 16; ++j2) v += soff[j][j2] * h1[j2];
    t[((size_t)b * 16 + j) * HIDD + o] = v;
    s += v; s2 += v * v;
  }
  atomicAdd(&stats[o], s);
  atomicAdd(&stats[128 + o], s2);
}

// BN apply (+relu/+residual), float4 vectorized.
__global__ void bn_apply_kernel(const float* __restrict__ t, const float* __restrict__ stats,
                                const float* __restrict__ g, const float* __restrict__ bt,
                                const float* __restrict__ res, float* __restrict__ out,
                                int do_relu) {
  int idx = (blockIdx.x * blockDim.x + threadIdx.x) * 4;
  if (idx >= NROWS * HIDD) return;
  int o = idx & 127;
  float inv = 1.0f / (float)NROWS;
  float4 tv = *(const float4*)&t[idx];
  float4 ov;
  float vv[4] = {tv.x, tv.y, tv.z, tv.w};
  float ro[4];
#pragma unroll
  for (int q = 0; q < 4; ++q) {
    int oc = o + q;
    float m = stats[oc] * inv;
    float var = stats[128 + oc] * inv - m * m;
    float v = (vv[q] - m) * rsqrtf(var + 1e-5f) * g[oc] + bt[oc];
    if (do_relu) v = fmaxf(v, 0.f);
    ro[q] = v;
  }
  if (res) {
    float4 rv = *(const float4*)&res[idx];
    ro[0] += rv.x; ro[1] += rv.y; ro[2] += rv.z; ro[3] += rv.w;
  }
  ov.x = ro[0]; ov.y = ro[1]; ov.z = ro[2]; ov.w = ro[3];
  *(float4*)&out[idx] = ov;
}

// ---------------- residual gconv: [diag*h | off@h] @ [W0;W1] + bias ----------------
// M=B*16, K=256, N=128. 128 threads = 4 waves; wave w owns batch blockIdx.x*4+w.
// A' written fragment-swizzled to LDS; B frags read straight from pre-swizzled global blob.
__global__ __launch_bounds__(128) void gconv_gemm_kernel(
    const float* __restrict__ h, const _Float16* __restrict__ wblob,
    const float* __restrict__ bias, const float* __restrict__ diag,
    const float* __restrict__ offm, float* __restrict__ t, float* __restrict__ stats) {
  __shared__ __align__(32) _Float16 aps[4][8][32][16];   // 32 KB swizzled A'
  __shared__ float soff[16][16];
  __shared__ float sdiag[16];
  __shared__ float sstat[256];
  int tid = threadIdx.x, lane = tid & 31, w = tid >> 5;
  int b = blockIdx.x * 4 + w;
  for (int i = tid; i < 256; i += 128) { soff[i >> 4][i & 15] = offm[i]; sstat[i] = 0.f; }
  if (tid < 16) sdiag[tid] = diag[tid];
  __syncthreads();

  const float* hb = h + (size_t)b * JJ * HIDD;
  __builtin_prefetch(hb + 4 * JJ * HIDD, 0, 1);  // global_prefetch_b8: next block's tile
#pragma unroll
  for (int cc = 0; cc < 2; ++cc) {
    int f = 2 * lane + cc * 64;                  // column pair (f, f+1)
    float c0[16], c1[16];
#pragma unroll
    for (int j = 0; j < 16; ++j) {
      float2 v = *(const float2*)&hb[j * HIDD + f];
      c0[j] = v.x; c1[j] = v.y;
    }
    int kk = f & 31, ks0 = f >> 5, lt, e;
    amap(kk, lt, e);
#pragma unroll
    for (int j = 0; j < 16; ++j) {
      float a1x = 0.f, a1y = 0.f;
#pragma unroll
      for (int j2 = 0; j2 < 16; ++j2) { a1x += soff[j][j2] * c0[j2]; a1y += soff[j][j2] * c1[j2]; }
      v2h p0; p0[0] = (_Float16)(sdiag[j] * c0[j]); p0[1] = (_Float16)(sdiag[j] * c1[j]);
      v2h p1; p1[0] = (_Float16)a1x; p1[1] = (_Float16)a1y;
      *(v2h*)&aps[w][ks0][lt + j][e]     = p0;   // A0 part (K = f)
      *(v2h*)&aps[w][ks0 + 4][lt + j][e] = p1;   // A1 part (K = 128 + f)
    }
  }

  v8f acc[8];
#pragma unroll
  for (int n = 0; n < 8; ++n)
#pragma unroll
    for (int r = 0; r < 8; ++r) acc[n][r] = 0.f;

#pragma unroll
  for (int ks = 0; ks < 8; ++ks) {
    v16h a = *(const v16h*)&aps[w][ks][lane][0];
#pragma unroll
    for (int n = 0; n < 8; ++n) {
      v16h bf = *(const v16h*)&wblob[(size_t)((ks * 8 + n) * 32 + lane) * 16];
      acc[n] = WMMA_F16(a, bf, acc[n]);
    }
  }

  int nli = lane & 15, mb = (lane >> 4) * 8;
  float* tb = t + (size_t)b * JJ * HIDD;
#pragma unroll
  for (int n = 0; n < 8; ++n) {
    int o = n * 16 + nli;
    float bo = bias[o], s = 0.f, s2 = 0.f;
#pragma unroll
    for (int r = 0; r < 8; ++r) {
      float v = acc[n][r] + bo;
      tb[(mb + r) * HIDD + o] = v;
      s += v; s2 += v * v;
    }
    atomicAdd(&sstat[o], s);
    atomicAdd(&sstat[128 + o], s2);
  }
  __syncthreads();
  for (int i = tid; i < 256; i += 128) atomicAdd(&stats[i], sstat[i]);
}

// ---------------- nonlocal g|t|p convs: (M=B*16, K=128, N=192) ----------------
__global__ __launch_bounds__(128) void nl_conv_gemm_kernel(
    const float* __restrict__ h, const int* __restrict__ grouped,
    const _Float16* __restrict__ wblob,
    const float* __restrict__ gb, const float* __restrict__ tbv, const float* __restrict__ pb,
    float* __restrict__ convout) {
  __shared__ __align__(32) _Float16 aps[4][4][32][16];   // 16 KB
  __shared__ int sg[16];
  int tid = threadIdx.x, lane = tid & 31, w = tid >> 5;
  int b = blockIdx.x * 4 + w;
  if (tid < 16) sg[tid] = grouped[tid];
  __syncthreads();
  const float* hb = h + (size_t)b * JJ * HIDD;
#pragma unroll
  for (int cc = 0; cc < 2; ++cc) {
    int f = 2 * lane + cc * 64;
    int kk = f & 31, ks0 = f >> 5, lt, e;
    amap(kk, lt, e);
#pragma unroll
    for (int j = 0; j < 16; ++j) {
      float2 v = *(const float2*)&hb[sg[j] * HIDD + f];
      v2h p; p[0] = (_Float16)v.x; p[1] = (_Float16)v.y;
      *(v2h*)&aps[w][ks0][lt + j][e] = p;
    }
  }
  v8f acc[12];
#pragma unroll
  for (int n = 0; n < 12; ++n)
#pragma unroll
    for (int r = 0; r < 8; ++r) acc[n][r] = 0.f;
#pragma unroll
  for (int ks = 0; ks < 4; ++ks) {
    v16h a = *(const v16h*)&aps[w][ks][lane][0];
#pragma unroll
    for (int n = 0; n < 12; ++n) {
      v16h bf = *(const v16h*)&wblob[(size_t)((ks * 12 + n) * 32 + lane) * 16];
      acc[n] = WMMA_F16(a, bf, acc[n]);
    }
  }
  int nli = lane & 15, mb = (lane >> 4) * 8;
  float* cb = convout + (size_t)b * JJ * 192;
#pragma unroll
  for (int n = 0; n < 12; ++n) {
    int oc = n * 16 + nli;
    float bo = (oc < 64) ? gb[oc] : (oc < 128) ? tbv[oc - 64] : pb[oc - 128];
#pragma unroll
    for (int r = 0; r < 8; ++r) cb[(mb + r) * 192 + oc] = acc[n][r] + bo;
  }
}

// nonlocal middle: pooling, t/p contraction, f map, y = f @ gx. One block per batch.
__global__ __launch_bounds__(128) void nl_mid_kernel(
    const float* __restrict__ convout, const float* __restrict__ cpw /*128*/,
    float* __restrict__ y) {
  __shared__ float sc[16][192];
  __shared__ float gx[64][8];
  __shared__ float fv[16][8];
  __shared__ float tv[16], pv[8];
  int b = blockIdx.x, tid = threadIdx.x;
  const float* cb = convout + (size_t)b * 16 * 192;
  for (int i = tid; i < 16 * 192; i += 128) sc[i / 192][i % 192] = cb[i];
  __syncthreads();
  for (int i = tid; i < 64 * 8; i += 128) {
    int c = i >> 3, wd = i & 7;
    gx[c][wd] = fmaxf(sc[2 * wd][c], sc[2 * wd + 1][c]);
  }
  if (tid < 16) {
    float s = 0.f;
    for (int c = 0; c < 64; ++c) s += sc[tid][64 + c] * cpw[c];
    tv[tid] = s;
  } else if (tid < 24) {
    int wd = tid - 16;
    float s = 0.f;
    for (int c = 0; c < 64; ++c)
      s += fmaxf(sc[2 * wd][128 + c], sc[2 * wd + 1][128 + c]) * cpw[64 + c];
    pv[wd] = s;
  }
  __syncthreads();
  if (tid < 128) {
    int hd = tid >> 3, wd = tid & 7;
    fv[hd][wd] = fmaxf(tv[hd] + pv[wd], 0.f) * (1.f / 8.f);
  }
  __syncthreads();
  float* yb = y + (size_t)b * 16 * 64;
  for (int i = tid; i < 16 * 64; i += 128) {
    int hd = i >> 6, c = i & 63;
    float s = 0.f;
#pragma unroll
    for (int wd = 0; wd < 8; ++wd) s += fv[hd][wd] * gx[c][wd];
    yb[hd * 64 + c] = s;
  }
}

// nonlocal Wy projection: (M=B*16, K=64, N=128) + BN stats.
__global__ __launch_bounds__(128) void nl_wy_gemm_kernel(
    const float* __restrict__ y, const _Float16* __restrict__ wblob,
    const float* __restrict__ bias, float* __restrict__ wy, float* __restrict__ stats) {
  __shared__ __align__(32) _Float16 aps[4][2][32][16];   // 8 KB
  __shared__ float sstat[256];
  int tid = threadIdx.x, lane = tid & 31, w = tid >> 5;
  int b = blockIdx.x * 4 + w;
  for (int i = tid; i < 256; i += 128) sstat[i] = 0.f;
  __syncthreads();
  const float* yb = y + (size_t)b * JJ * INTERD;
  {
    int f = 2 * lane;                       // covers K = 0..63 as pairs
    int kk = f & 31, ks0 = f >> 5, lt, e;
    amap(kk, lt, e);
#pragma unroll
    for (int j = 0; j < 16; ++j) {
      float2 v = *(const float2*)&yb[j * INTERD + f];
      v2h p; p[0] = (_Float16)v.x; p[1] = (_Float16)v.y;
      *(v2h*)&aps[w][ks0][lt + j][e] = p;
    }
  }
  v8f acc[8];
#pragma unroll
  for (int n = 0; n < 8; ++n)
#pragma unroll
    for (int r = 0; r < 8; ++r) acc[n][r] = 0.f;
#pragma unroll
  for (int ks = 0; ks < 2; ++ks) {
    v16h a = *(const v16h*)&aps[w][ks][lane][0];
#pragma unroll
    for (int n = 0; n < 8; ++n) {
      v16h bf = *(const v16h*)&wblob[(size_t)((ks * 8 + n) * 32 + lane) * 16];
      acc[n] = WMMA_F16(a, bf, acc[n]);
    }
  }
  int nli = lane & 15, mb = (lane >> 4) * 8;
  float* ob = wy + (size_t)b * JJ * HIDD;
#pragma unroll
  for (int n = 0; n < 8; ++n) {
    int o = n * 16 + nli;
    float bo = bias[o], s = 0.f, s2 = 0.f;
#pragma unroll
    for (int r = 0; r < 8; ++r) {
      float v = acc[n][r] + bo;
      ob[(mb + r) * HIDD + o] = v;
      s += v; s2 += v * v;
    }
    atomicAdd(&sstat[o], s);
    atomicAdd(&sstat[128 + o], s2);
  }
  __syncthreads();
  for (int i = tid; i < 256; i += 128) atomicAdd(&stats[i], sstat[i]);
}

// nonlocal finalize: out[b,j,o] = bn(Wy[b, restored[j], o]) + h[b,j,o], float4.
__global__ void nl_finalize_kernel(const float* __restrict__ wy, const float* __restrict__ stats,
                                   const float* __restrict__ g, const float* __restrict__ bt,
                                   const int* __restrict__ restored, float* __restrict__ hbuf) {
  int idx = (blockIdx.x * blockDim.x + threadIdx.x) * 4;
  if (idx >= NROWS * HIDD) return;
  int o = idx & 127;
  int row = idx >> 7;
  int b = row >> 4, j = row & 15;
  int gp = restored[j];
  float inv = 1.f / (float)NROWS;
  float4 wv = *(const float4*)&wy[((size_t)(b * 16 + gp)) * HIDD + o];
  float4 hv = *(const float4*)&hbuf[idx];
  float wq[4] = {wv.x, wv.y, wv.z, wv.w};
  float hq[4] = {hv.x, hv.y, hv.z, hv.w};
  float rq[4];
#pragma unroll
  for (int q = 0; q < 4; ++q) {
    int oc = o + q;
    float m = stats[oc] * inv;
    float var = stats[128 + oc] * inv - m * m;
    rq[q] = (wq[q] - m) * rsqrtf(var + 1e-5f) * g[oc] + bt[oc] + hq[q];
  }
  float4 ov; ov.x = rq[0]; ov.y = rq[1]; ov.z = rq[2]; ov.w = rq[3];
  *(float4*)&hbuf[idx] = ov;
}

// final gconv: N=3 output, VALU.
__global__ __launch_bounds__(64) void gc_out_kernel(
    const float* __restrict__ h, const float* __restrict__ W /*(2,128,3)*/,
    const float* __restrict__ bias, const float* __restrict__ diag,
    const float* __restrict__ offm, float* __restrict__ out) {
  __shared__ float hs[16][HIDD];
  __shared__ float soff[16][16];
  __shared__ float sdiag[16];
  int b = blockIdx.x, tid = threadIdx.x;
  for (int i = tid; i < 16 * HIDD; i += 64) hs[i >> 7][i & 127] = h[(size_t)b * 2048 + i];
  for (int i = tid; i < 256; i += 64) soff[i >> 4][i & 15] = offm[i];
  if (tid < 16) sdiag[tid] = diag[tid];
  __syncthreads();
  if (tid < 48) {
    int j = tid / 3, o = tid % 3;
    float s = bias[o];
    for (int f = 0; f < 128; ++f) {
      float g1 = 0.f;
#pragma unroll
      for (int j2 = 0; j2 < 16; ++j2) g1 += soff[j][j2] * hs[j2][f];
      s += sdiag[j] * hs[j][f] * W[f * 3 + o] + g1 * W[384 + f * 3 + o];
    }
    out[(size_t)b * 48 + tid] = s;
  }
}

// ---------------- host-side pipeline ----------------
extern "C" void kernel_launch(void* const* d_in, const int* in_sizes, int n_in,
                              void* d_out, int out_size, void* d_ws, size_t ws_size,
                              hipStream_t stream) {
  const float* x        = (const float*)d_in[0];
  const float* gc_in_W  = (const float*)d_in[1];
  const float* gc_in_e  = (const float*)d_in[2];
  const float* gc_in_b  = (const float*)d_in[3];
  const float* bn_in_g  = (const float*)d_in[4];
  const float* bn_in_b  = (const float*)d_in[5];
  const float* W_res    = (const float*)d_in[6];
  const float* e_res    = (const float*)d_in[7];
  const float* b_res    = (const float*)d_in[8];
  const float* bng_res  = (const float*)d_in[9];
  const float* bnb_res  = (const float*)d_in[10];
  const float* nl_g_w   = (const float*)d_in[11];
  const float* nl_g_b   = (const float*)d_in[12];
  const float* nl_t_w   = (const float*)d_in[13];
  const float* nl_t_b   = (const float*)d_in[14];
  const float* nl_p_w   = (const float*)d_in[15];
  const float* nl_p_b   = (const float*)d_in[16];
  const float* nl_cp_w  = (const float*)d_in[17];
  const float* nl_W_w   = (const float*)d_in[18];
  const float* nl_W_b   = (const float*)d_in[19];
  const float* nl_bn_g  = (const float*)d_in[20];
  const float* nl_bn_b  = (const float*)d_in[21];
  const float* gc_out_W = (const float*)d_in[22];
  const float* gc_out_e = (const float*)d_in[23];
  const float* gc_out_b = (const float*)d_in[24];
  const int*   mrows    = (const int*)d_in[25];
  const int*   mcols    = (const int*)d_in[26];
  const int*   grouped  = (const int*)d_in[27];
  const int*   restored = (const int*)d_in[28];
  int nnz = in_sizes[2];

  float* ws    = (float*)d_ws;
  float* stats = ws + OFF_STATS;
  float* diag  = ws + OFF_DIAG;
  float* offm  = ws + OFF_OFFM;
  float* H   = ws + OFF_H;
  float* T   = ws + OFF_T;
  float* H2  = ws + OFF_H2;
  float* C   = ws + OFF_C;
  float* Y   = ws + OFF_Y;
  float* WY  = ws + OFF_WY;
  _Float16* wf = (_Float16*)(ws + OFF_F16);
  _Float16* WG = wf + WBLOB_GCONV;
  _Float16* WC = wf + WBLOB_NLCONV;
  _Float16* WW = wf + WBLOB_NLWY;
  float* out = (float*)d_out;

  const int EW_BLOCKS = (NROWS * HIDD) / (256 * 4);  // 8192 (float4 elementwise)

  zero_kernel<<<14, 256, 0, stream>>>(stats, 14 * 256);
  prep_adj_kernel<<<1, 192, 0, stream>>>(gc_in_e, e_res, gc_out_e, mrows, mcols, nnz,
                                         diag, offm);
  swz_gconv_w_kernel<<<dim3(128, 8), 256, 0, stream>>>(W_res, WG);
  swz_nlconv_w_kernel<<<dim3(96, 5), 256, 0, stream>>>(nl_g_w, nl_t_w, nl_p_w, WC);
  swz_nlwy_w_kernel<<<dim3(32, 5), 256, 0, stream>>>(nl_W_w, WW);

  // input graph_conv
  gc_in_kernel<<<BB, 128, 0, stream>>>(x, gc_in_W, gc_in_b, diag, offm, T, stats);
  bn_apply_kernel<<<EW_BLOCKS, 256, 0, stream>>>(T, stats, bn_in_g, bn_in_b, nullptr, H, 1);

  auto nonlocal = [&](int i) {
    nl_conv_gemm_kernel<<<BB / 4, 128, 0, stream>>>(
        H, grouped, WC + (size_t)i * 24576,
        nl_g_b + (size_t)i * INTERD, nl_t_b + (size_t)i * INTERD, nl_p_b + (size_t)i * INTERD, C);
    nl_mid_kernel<<<BB, 128, 0, stream>>>(C, nl_cp_w + (size_t)i * 2 * INTERD, Y);
    nl_wy_gemm_kernel<<<BB / 4, 128, 0, stream>>>(
        Y, WW + (size_t)i * 8192, nl_W_b + (size_t)i * HIDD,
        WY, stats + (size_t)(9 + i) * 256);
    nl_finalize_kernel<<<EW_BLOCKS, 256, 0, stream>>>(
        WY, stats + (size_t)(9 + i) * 256,
        nl_bn_g + (size_t)i * HIDD, nl_bn_b + (size_t)i * HIDD, restored, H);
  };

  nonlocal(0);
  for (int i = 0; i < NLAY; ++i) {
    int l0 = 2 * i, l1 = 2 * i + 1;
    gconv_gemm_kernel<<<BB / 4, 128, 0, stream>>>(
        H, WG + (size_t)l0 * 32768, b_res + (size_t)l0 * HIDD,
        diag + (size_t)(1 + l0) * 16, offm + (size_t)(1 + l0) * 256,
        T, stats + (size_t)(1 + l0) * 256);
    bn_apply_kernel<<<EW_BLOCKS, 256, 0, stream>>>(
        T, stats + (size_t)(1 + l0) * 256,
        bng_res + (size_t)l0 * HIDD, bnb_res + (size_t)l0 * HIDD, nullptr, H2, 1);
    gconv_gemm_kernel<<<BB / 4, 128, 0, stream>>>(
        H2, WG + (size_t)l1 * 32768, b_res + (size_t)l1 * HIDD,
        diag + (size_t)(1 + l1) * 16, offm + (size_t)(1 + l1) * 256,
        T, stats + (size_t)(1 + l1) * 256);
    bn_apply_kernel<<<EW_BLOCKS, 256, 0, stream>>>(
        T, stats + (size_t)(1 + l1) * 256,
        bng_res + (size_t)l1 * HIDD, bnb_res + (size_t)l1 * HIDD, H /*residual*/, H, 1);
    nonlocal(i + 1);
  }
  gc_out_kernel<<<BB, 64, 0, stream>>>(H, gc_out_W, gc_out_b,
                                       diag + 9 * 16, offm + 9 * 256, out);
}
